// Attention_49838800503101
// MI455X (gfx1250) — matrix operand order repaired and checked by reading, not compiled, not run on hardware
//
#include <hip/hip_runtime.h>

#define B_    2
#define S_    4096
#define D_    768
#define H_    12
#define HD_   64
#define MB_   24
#define NBR_  128
#define NSEL_ 1024

typedef __attribute__((ext_vector_type(16))) _Float16 v16h;
typedef __attribute__((ext_vector_type(8)))  _Float16 v8h;
typedef __attribute__((ext_vector_type(4)))  _Float16 v4h;
typedef __attribute__((ext_vector_type(8)))  float    v8f;

// ---------------- CDNA5 async global->LDS copy (signature from probe) -------
// Probe round 2 revealed: param type is `vector_size(4*sizeof(int)) int *`.
typedef int v4i_b128 __attribute__((vector_size(4 * sizeof(int))));
typedef __attribute__((address_space(1))) v4i_b128 gas_v4i;
typedef __attribute__((address_space(3))) v4i_b128 las_v4i;

#if defined(__gfx1250__) && __has_builtin(__builtin_amdgcn_global_load_async_to_lds_b128)
#define HAVE_ASYNC 1
#else
#define HAVE_ASYNC 0
#endif

#if HAVE_ASYNC
__device__ __forceinline__ void async_b128(const void* g, void* l) {
  // emits global_load_async_to_lds_b128 (ASYNCcnt-tracked DMA into LDS)
  __builtin_amdgcn_global_load_async_to_lds_b128((gas_v4i*)g, (las_v4i*)l, 0, 0);
}
#if __has_builtin(__builtin_amdgcn_s_wait_asynccnt)
#define WAIT_ASYNC(n) __builtin_amdgcn_s_wait_asynccnt(n)
#else
#define WAIT_ASYNC(n) asm volatile("s_wait_asynccnt " #n ::: "memory")
#endif
#else
__device__ __forceinline__ void async_b128(const void* g, void* l) {
  *(v8h*)l = *(const v8h*)g;   // synchronous fallback
}
#define WAIT_ASYNC(n) ((void)0)
#endif

__device__ __forceinline__ v8f wmma16(v16h a, v16h b, v8f c) {
  // emits v_wmma_f32_16x16x32_f16
  return __builtin_amdgcn_wmma_f32_16x16x32_f16(false, a, false, b, (short)0, c,
                                                false, false);
}

__device__ __forceinline__ v8f splat8(float x) {
  v8f r;
#pragma unroll
  for (int i = 0; i < 8; ++i) r[i] = x;
  return r;
}
__device__ __forceinline__ v8f vmax8(v8f a, v8f b) {
  v8f r;
#pragma unroll
  for (int i = 0; i < 8; ++i) r[i] = fmaxf(a[i], b[i]);
  return r;
}
__device__ __forceinline__ v8f shflxor8(v8f x, int m) {
  v8f r;
#pragma unroll
  for (int i = 0; i < 8; ++i) r[i] = __shfl_xor(x[i], m, 32);
  return r;
}

// A-operand fragment (16xK tile, row-major LDS, K contiguous per row).
// lane&15 = row M; lane half selects K chunks {ko..ko+7, 16+ko..16+ko+7}, ko=0/8.
__device__ __forceinline__ v16h ldfragA(const _Float16* p0, int stride, int lane) {
  int m  = lane & 15;
  int ko = (lane & 16) ? 8 : 0;
  const _Float16* p = p0 + m * stride + ko;
  v8h lo = *(const v8h*)(p);
  v8h hi = *(const v8h*)(p + 16);
  return __builtin_shufflevector(lo, hi, 0, 1, 2, 3, 4, 5, 6, 7,
                                 8, 9, 10, 11, 12, 13, 14, 15);
}
// B-operand fragment (Kx16 tile stored transposed: row = N, K contiguous).
// lane&15 = column N; lane half selects K chunks {ko..ko+7, 16+ko..16+ko+7}, ko=0/8.
__device__ __forceinline__ v16h ldfragB(const _Float16* p0, int stride, int lane) {
  int n  = lane & 15;
  int ko = (lane & 16) ? 8 : 0;
  const _Float16* p = p0 + n * stride + ko;
  v8h lo = *(const v8h*)(p);
  v8h hi = *(const v8h*)(p + 16);
  return __builtin_shufflevector(lo, hi, 0, 1, 2, 3, 4, 5, 6, 7,
                                 8, 9, 10, 11, 12, 13, 14, 15);
}

__device__ __forceinline__ unsigned f2key(float f) {
  int i = __float_as_int(f);
  return (i >= 0) ? ((unsigned)i | 0x80000000u) : ~(unsigned)i;
}
__device__ __forceinline__ float key2f(unsigned u) {
  int i = (u & 0x80000000u) ? (int)(u & 0x7fffffffu) : ~(int)u;
  return __int_as_float(i);
}

// ---------------------------------------------------------------------------
// Kernel 0: bulk f32 -> f16 conversion (X and the three weight matrices), so
// the GEMM K-loop is pure async-copy + WMMA with no VALU converts.
// ---------------------------------------------------------------------------
__global__ void __launch_bounds__(256)
cvt_f32_f16_kernel(const float* __restrict__ src, _Float16* __restrict__ dst,
                   int n4) {
  int i = blockIdx.x * 256 + threadIdx.x;
  if (i < n4) {
    float4 x = ((const float4*)src)[i];
    v4h h;
    h[0] = (_Float16)x.x; h[1] = (_Float16)x.y;
    h[2] = (_Float16)x.z; h[3] = (_Float16)x.w;
    ((v4h*)dst)[i] = h;
  }
}

// ---------------------------------------------------------------------------
// Kernel 1: QKV projection. 128x64 tile per WG (8 waves, 2x2 WMMA tiles/wave),
// K-step 64, double-buffered LDS tiles filled by async global->LDS copies.
// fp16 output in [MB, S, 64] head-major layout, pre-multiplied by the mask.
// ---------------------------------------------------------------------------
__global__ void __launch_bounds__(256)
qkv_proj_kernel(const _Float16* __restrict__ Xh, const _Float16* __restrict__ Wh,
                const float* __restrict__ mask,
                const float* __restrict__ bq, const float* __restrict__ bk,
                const float* __restrict__ bv,
                _Float16* __restrict__ Qf, _Float16* __restrict__ Kf,
                _Float16* __restrict__ Vf) {
  __shared__ __align__(16) _Float16 As[2][128 * 72];
  __shared__ __align__(16) _Float16 Bs[2][64 * 72];
  const int t = threadIdx.x, lane = t & 31, w = t >> 5;
  const int wm = w & 3, wn = w >> 2;
  const int row0 = blockIdx.x * 128;
  const int h = blockIdx.y, z = blockIdx.z;
  const _Float16* Wsel = Wh + (size_t)z * D_ * D_;
  const float* bi = (z == 0) ? bq : (z == 1) ? bk : bv;
  _Float16* Out   = (z == 0) ? Qf : (z == 1) ? Kf : Vf;
  const int e0 = h * 64;

  v8f acc[2][2];
  acc[0][0] = splat8(0.f); acc[0][1] = splat8(0.f);
  acc[1][0] = splat8(0.f); acc[1][1] = splat8(0.f);

  auto stage = [&](int ki, int buf) {
    const int kk = ki * 64;
    // A tile: 128 rows x 64 halfs = 1024 16B chunks, 4 per thread
#pragma unroll
    for (int j = 0; j < 4; ++j) {
      int id = j * 256 + t;
      int row = id >> 3, c8 = (id & 7) * 8;
      async_b128(Xh + (size_t)(row0 + row) * D_ + kk + c8,
                 &As[buf][row * 72 + c8]);
    }
    // B tile: 64 rows x 64 halfs = 512 chunks, 2 per thread
#pragma unroll
    for (int j = 0; j < 2; ++j) {
      int id = j * 256 + t;
      int row = id >> 3, c8 = (id & 7) * 8;
      async_b128(Wsel + (size_t)(e0 + row) * D_ + kk + c8,
                 &Bs[buf][row * 72 + c8]);
    }
  };

  stage(0, 0);
  for (int ki = 0; ki < 12; ++ki) {
    const int buf = ki & 1;
    if (ki + 1 < 12) {
      stage(ki + 1, buf ^ 1);   // DMA next tiles while we compute
      WAIT_ASYNC(6);            // 6 async instrs/wave per stage, in-order
    } else {
      WAIT_ASYNC(0);
    }
    __syncthreads();
#pragma unroll
    for (int kk2 = 0; kk2 < 64; kk2 += 32) {
      v16h a0 = ldfragA(&As[buf][(wm * 32     ) * 72 + kk2], 72, lane);
      v16h a1 = ldfragA(&As[buf][(wm * 32 + 16) * 72 + kk2], 72, lane);
      v16h b0 = ldfragB(&Bs[buf][(wn * 32     ) * 72 + kk2], 72, lane);
      v16h b1 = ldfragB(&Bs[buf][(wn * 32 + 16) * 72 + kk2], 72, lane);
      acc[0][0] = wmma16(a0, b0, acc[0][0]);
      acc[0][1] = wmma16(a0, b1, acc[0][1]);
      acc[1][0] = wmma16(a1, b0, acc[1][0]);
      acc[1][1] = wmma16(a1, b1, acc[1][1]);
    }
    __syncthreads();
  }

  const int hl = lane >> 4, ln = lane & 15;
#pragma unroll
  for (int mi = 0; mi < 2; ++mi)
#pragma unroll
    for (int ni = 0; ni < 2; ++ni) {
      int n = wn * 32 + ni * 16 + ln;
      float bval = bi[e0 + n];
#pragma unroll
      for (int r = 0; r < 8; ++r) {
        int row = row0 + wm * 32 + mi * 16 + r + 8 * hl;
        int bb = row >> 12, s = row & 4095;
        float mv = mask[(size_t)bb * S_ + s];
        Out[((size_t)(bb * H_ + h) * S_ + s) * HD_ + n] =
            (_Float16)((acc[mi][ni][r] + bval) * mv);
      }
    }
}

// ---------------------------------------------------------------------------
// Kernel 2: per-block means Qh/Kh/Vh and token counts.
// ---------------------------------------------------------------------------
__global__ void __launch_bounds__(64)
block_mean_kernel(const _Float16* __restrict__ Qf, const _Float16* __restrict__ Kf,
                  const _Float16* __restrict__ Vf, const float* __restrict__ mask,
                  float* __restrict__ Qh, float* __restrict__ Kh,
                  float* __restrict__ Vh, float* __restrict__ tc) {
  int blk = blockIdx.x;
  int mb = blk >> 7, n = blk & 127;
  int b = mb / H_;
  int d = threadIdx.x;
  size_t base = ((size_t)mb * S_ + n * 32) * HD_;
  const float* mrow = mask + (size_t)b * S_ + n * 32;
  float sq = 0.f, sk = 0.f, sv = 0.f, cnt = 0.f;
  for (int t = 0; t < 32; ++t) {
    cnt += mrow[t];  // Q/K/V already mask-multiplied at projection store
    size_t o = base + (size_t)t * HD_ + d;
    sq += (float)Qf[o]; sk += (float)Kf[o]; sv += (float)Vf[o];
  }
  float inv = 1.f / (cnt + 1e-6f);
  size_t ob = (size_t)(mb * NBR_ + n) * HD_ + d;
  Qh[ob] = sq * inv; Kh[ob] = sk * inv; Vh[ob] = sv * inv;
  if (d == 0) tc[mb * NBR_ + n] = cnt;
}

// ---------------------------------------------------------------------------
// Kernel 3: one WG per head: 128x128 low-res logits via WMMA, row max, prior
// (empty-pair mask + diagonal band), exact radix-select of the 1024th largest
// prior (monotonic float->u32 keys), row-bucketed selected-block lists.
// ---------------------------------------------------------------------------
__global__ void __launch_bounds__(256)
lowres_select_kernel(const float* __restrict__ Qh, const float* __restrict__ Kh,
                     const float* __restrict__ tc, float* __restrict__ ll_g,
                     float* __restrict__ rm_g, float* __restrict__ th_g,
                     int* __restrict__ rcnt_g, int* __restrict__ rofs_g,
                     int* __restrict__ sel_g) {
  extern __shared__ char smem3[];
  _Float16* Qs   = (_Float16*)smem3;            // 128*72 fp16
  _Float16* Ks   = Qs + 128 * 72;               // 128*72 fp16
  float*    lowS = (float*)(Ks + 128 * 72);     // 128*129 f32 (reused as keys)
  unsigned* keyS = (unsigned*)lowS;
  float*    rmaxS = lowS + 128 * 129;           // 128
  int* irow = (int*)(rmaxS + 128);              // 128
  int* iofs = irow + 128;                       // 128
  int* icur = iofs + 128;                       // 128
  int* scan = icur + 128;                       // 256
  int* red  = scan + 256;                       // 1

  const int mb = blockIdx.x;
  const int t = threadIdx.x, lane = t & 31, w = t >> 5;

  for (int i = t; i < 128 * 64; i += 256) {
    int r = i >> 6, c = i & 63;
    Qs[r * 72 + c] = (_Float16)Qh[(size_t)mb * NBR_ * HD_ + i];
    Ks[r * 72 + c] = (_Float16)Kh[(size_t)mb * NBR_ * HD_ + i];
  }
  __syncthreads();

  // wave w: rows [w*16, w*16+16) x all 128 cols, K=64 (2 WMMA k-steps)
  v8f acc[8];
#pragma unroll
  for (int nt = 0; nt < 8; ++nt) acc[nt] = splat8(0.f);
  for (int kk = 0; kk < 64; kk += 32) {
    v16h a = ldfragA(Qs + (w * 16) * 72 + kk, 72, lane);
#pragma unroll
    for (int nt = 0; nt < 8; ++nt) {
      v16h b = ldfragB(Ks + (nt * 16) * 72 + kk, 72, lane);
      acc[nt] = wmma16(a, b, acc[nt]);
    }
  }
  const int hl = lane >> 4, ln = lane & 15;
#pragma unroll
  for (int nt = 0; nt < 8; ++nt)
#pragma unroll
    for (int r = 0; r < 8; ++r)
      lowS[(w * 16 + r + 8 * hl) * 129 + nt * 16 + ln] = acc[nt][r] * 0.125f;
  __syncthreads();

  if (t < 128) {                 // row max on RAW logits (reference order)
    float mx = -1e30f;
    for (int c = 0; c < 128; ++c) mx = fmaxf(mx, lowS[t * 129 + c]);
    rmaxS[t] = mx;
    rm_g[mb * NBR_ + t] = mx;
  }
  __syncthreads();

  // empty-pair masking, store low_logit, overwrite LDS with prior keys
  for (int j = 0; j < 64; ++j) {
    int fi = t * 64 + j;
    int r = fi >> 7, c = fi & 127;
    float raw = lowS[r * 129 + c];
    float emp = (tc[mb * NBR_ + r] * tc[mb * NBR_ + c] < 0.5f) ? 1.f : 0.f;
    float msk = raw - 1e4f * emp;
    ll_g[(size_t)mb * NBR_ * NBR_ + fi] = msk;
    int dd = r - c; if (dd < 0) dd = -dd;
    float band = (dd <= 1) ? 5e3f : 0.f;     // DIAG_N=3 -> off=1
    float prior = msk - rmaxS[r] + band;
    keyS[r * 129 + c] = f2key(prior);
  }
  __syncthreads();

  // radix-select the 1024th-largest key (exact)
  unsigned cur = 0u;
  for (int bit = 31; bit >= 0; --bit) {
    unsigned cand = cur | (1u << bit);
    if (t == 0) red[0] = 0;
    __syncthreads();
    int loc = 0;
    for (int j = 0; j < 64; ++j) {
      int fi = t * 64 + j;
      loc += (keyS[(fi >> 7) * 129 + (fi & 127)] >= cand) ? 1 : 0;
    }
    atomicAdd(&red[0], loc);
    __syncthreads();
    int tot = red[0];
    __syncthreads();
    if (tot >= NSEL_) cur = cand;
  }
  if (t == 0) { red[0] = 0; th_g[mb] = key2f(cur); }
  __syncthreads();
  {
    int loc = 0;
    for (int j = 0; j < 64; ++j) {
      int fi = t * 64 + j;
      loc += (keyS[(fi >> 7) * 129 + (fi & 127)] > cur) ? 1 : 0;
    }
    atomicAdd(&red[0], loc);
  }
  __syncthreads();
  const int c1 = red[0];           // strictly greater than threshold
  const int rt = NSEL_ - c1;       // ties to accept, in flat-index order
  __syncthreads();

  int tie = 0;
  for (int j = 0; j < 64; ++j) {
    int fi = t * 64 + j;
    tie += (keyS[(fi >> 7) * 129 + (fi & 127)] == cur) ? 1 : 0;
  }
  scan[t] = tie;
  __syncthreads();
  for (int off = 1; off < 256; off <<= 1) {
    int v = (t >= off) ? scan[t - off] : 0;
    __syncthreads();
    scan[t] += v;
    __syncthreads();
  }
  const int tiebase = scan[t] - tie;

  if (t < 128) { irow[t] = 0; icur[t] = 0; }
  __syncthreads();
  {  // pass 1: per-row counts
    int lt = 0;
    for (int j = 0; j < 64; ++j) {
      int fi = t * 64 + j;
      int r = fi >> 7, c = fi & 127;
      unsigned k = keyS[r * 129 + c];
      bool selb = (k > cur);
      if (!selb && k == cur) { selb = (tiebase + lt) < rt; ++lt; }
      if (selb) atomicAdd(&irow[r], 1);
    }
  }
  __syncthreads();
  if (t == 0) {
    int o = 0;
    for (int r = 0; r < 128; ++r) { iofs[r] = o; o += irow[r]; }
  }
  __syncthreads();
  {  // pass 2: scatter column indices into row buckets
    int lt = 0;
    for (int j = 0; j < 64; ++j) {
      int fi = t * 64 + j;
      int r = fi >> 7, c = fi & 127;
      unsigned k = keyS[r * 129 + c];
      bool selb = (k > cur);
      if (!selb && k == cur) { selb = (tiebase + lt) < rt; ++lt; }
      if (selb) {
        int pos = iofs[r] + atomicAdd(&icur[r], 1);
        if (pos < NSEL_) sel_g[mb * NSEL_ + pos] = c;
      }
    }
  }
  __syncthreads();
  if (t < 128) {
    rcnt_g[mb * NBR_ + t] = irow[t];
    rofs_g[mb * NBR_ + t] = iofs[t];
  }
}

// ---------------------------------------------------------------------------
// Kernel 4: one wave32 per (head, query block). Flash-style online softmax
// over its selected key blocks, with async double-buffered K/V staging,
// fused with the low-resolution branch, corrections, and final normalization.
// ---------------------------------------------------------------------------
__global__ void __launch_bounds__(32)
mra_attn_kernel(const _Float16* __restrict__ Qf, const _Float16* __restrict__ Kf,
                const _Float16* __restrict__ Vf, const float* __restrict__ mask,
                const float* __restrict__ Vh, const float* __restrict__ tc,
                const float* __restrict__ ll, const float* __restrict__ rm,
                const float* __restrict__ th_g, const int* __restrict__ rcnt,
                const int* __restrict__ rofs, const int* __restrict__ sel,
                float* __restrict__ out) {
  __shared__ __align__(16) _Float16 Qs[32 * 72];
  __shared__ __align__(16) _Float16 Ks2[2][32 * 72];
  __shared__ __align__(16) _Float16 Vs2[2][32 * 64];  // row-major V (DMA target)
  __shared__ __align__(16) _Float16 Vt[64 * 40];      // V transposed: [d][token]
  __shared__ __align__(16) _Float16 Ps[32 * 40];      // P for A-fragment reads
  __shared__ float lowW[128];
  __shared__ float lowO[64];

  const int rblk = blockIdx.x, mb = blockIdx.y;
  const int lane = threadIdx.x, hl = lane >> 4, ln = lane & 15;
  const int b = mb / H_, h = mb % H_;

  {
    const v8h* src = (const v8h*)(Qf + ((size_t)mb * S_ + rblk * 32 + lane) * HD_);
    v8h* dst = (v8h*)(Qs + lane * 72);
#pragma unroll
    for (int j = 0; j < 8; ++j) dst[j] = src[j];
  }

  v8f O[2][4], mrun[2], lrun[2];
#pragma unroll
  for (int mt = 0; mt < 2; ++mt) {
    mrun[mt] = splat8(-1e6f);   // matches reference's -1e6 empty-row guard
    lrun[mt] = splat8(0.f);
#pragma unroll
    for (int vt = 0; vt < 4; ++vt) O[mt][vt] = splat8(0.f);
  }

  const int cnt = rcnt[mb * NBR_ + rblk];
  const int* list = sel + mb * NSEL_ + rofs[mb * NBR_ + rblk];

  // 16 async instrs per staged block pair (8 for K tile, 8 for V tile)
  auto stage4 = [&](int idx, int buf) {
    const int cb = list[idx];
    const _Float16* kb = Kf + ((size_t)mb * S_ + cb * 32) * HD_;
    const _Float16* vb = Vf + ((size_t)mb * S_ + cb * 32) * HD_;
#pragma unroll
    for (int j = 0; j < 8; ++j) {
      int id = j * 32 + lane;
      int row = id >> 3, c8 = (id & 7) * 8;
      async_b128(kb + id * 8, &Ks2[buf][row * 72 + c8]);
      async_b128(vb + id * 8, &Vs2[buf][id * 8]);
    }
  };

  if (cnt > 0) stage4(0, 0);
  for (int it = 0; it < cnt; ++it) {
    const int buf = it & 1;
    if (it + 1 < cnt) {
      stage4(it + 1, buf ^ 1);  // DMA next block while we use this one
      WAIT_ASYNC(16);
    } else {
      WAIT_ASYNC(0);
    }
    const int cb = list[it];
    // transpose V (LDS -> LDS) for contiguous-K B-fragments
    {
      const v8h* vr = (const v8h*)&Vs2[buf][lane * 64];
#pragma unroll
      for (int j = 0; j < 8; ++j) {
        v8h x = vr[j];
#pragma unroll
        for (int e = 0; e < 8; ++e) Vt[(j * 8 + e) * 40 + lane] = x[e];
      }
    }
    float km[2];
    km[0] = mask[(size_t)b * S_ + cb * 32 + ln];
    km[1] = mask[(size_t)b * S_ + cb * 32 + 16 + ln];
    __syncthreads();

    v8f Sl[2][2];
#pragma unroll
    for (int mt = 0; mt < 2; ++mt)
#pragma unroll
      for (int nt = 0; nt < 2; ++nt) Sl[mt][nt] = splat8(0.f);
#pragma unroll
    for (int kk = 0; kk < 64; kk += 32) {
#pragma unroll
      for (int mt = 0; mt < 2; ++mt) {
        v16h a = ldfragA(Qs + (mt * 16) * 72 + kk, 72, lane);
#pragma unroll
        for (int nt = 0; nt < 2; ++nt) {
          v16h bf = ldfragB(&Ks2[buf][(nt * 16) * 72 + kk], 72, lane);
          Sl[mt][nt] = wmma16(a, bf, Sl[mt][nt]);
        }
      }
    }
#pragma unroll
    for (int mt = 0; mt < 2; ++mt) {
#pragma unroll
      for (int nt = 0; nt < 2; ++nt)
#pragma unroll
        for (int r = 0; r < 8; ++r) Sl[mt][nt][r] *= 0.125f;

      // raw-logit row max within block (pre key-mask, as in reference)
      v8f bm = vmax8(Sl[mt][0], Sl[mt][1]);
#pragma unroll
      for (int off = 1; off < 16; off <<= 1) bm = vmax8(bm, shflxor8(bm, off));
      v8f mnew = vmax8(mrun[mt], bm);
      v8f sc;
#pragma unroll
      for (int r = 0; r < 8; ++r) sc[r] = __expf(mrun[mt][r] - mnew[r]);
#pragma unroll
      for (int vt = 0; vt < 4; ++vt)
#pragma unroll
        for (int r = 0; r < 8; ++r) O[mt][vt][r] *= sc[r];
#pragma unroll
      for (int r = 0; r < 8; ++r) lrun[mt][r] *= sc[r];
      mrun[mt] = mnew;

      v8f rs = splat8(0.f);
#pragma unroll
      for (int nt = 0; nt < 2; ++nt) {
        float pen = -1e4f * (1.f - km[nt]);
#pragma unroll
        for (int r = 0; r < 8; ++r) {
          float p = __expf(Sl[mt][nt][r] + pen - mnew[r]);
          Ps[(mt * 16 + r + 8 * hl) * 40 + nt * 16 + ln] = (_Float16)p;
          rs[r] += p;
        }
      }
#pragma unroll
      for (int off = 1; off < 16; off <<= 1) {
        v8f o2 = shflxor8(rs, off);
#pragma unroll
        for (int r = 0; r < 8; ++r) rs[r] += o2[r];
      }
#pragma unroll
      for (int r = 0; r < 8; ++r) lrun[mt][r] += rs[r];
    }
    __syncthreads();
#pragma unroll
    for (int mt = 0; mt < 2; ++mt) {
      v16h a = ldfragA(Ps + (mt * 16) * 40, 40, lane);
#pragma unroll
      for (int vt = 0; vt < 4; ++vt) {
        v16h bf = ldfragB(Vt + (vt * 16) * 40, 40, lane);
        O[mt][vt] = wmma16(a, bf, O[mt][vt]);
      }
    }
    __syncthreads();
  }

  // ---- low-resolution branch (identical for all 32 query tokens of the row) ----
  const float rmax = rm[mb * NBR_ + rblk];
  const float thv = th_g[mb];
  float lsum = 0.f;
#pragma unroll
  for (int jj = 0; jj < 4; ++jj) {
    int c = lane + jj * 32;
    float l0 = ll[(size_t)mb * NBR_ * NBR_ + rblk * NBR_ + c];
    int dd = rblk - c; if (dd < 0) dd = -dd;
    float prior = l0 - rmax + ((dd <= 1) ? 5e3f : 0.f);
    float smask = (prior >= thv) ? 1.f : 0.f;
    float wv = __expf(l0 - rmax - 1e4f * smask) * tc[mb * NBR_ + c];
    lowW[c] = wv;
    lsum += wv;
  }
#pragma unroll
  for (int off = 1; off < 32; off <<= 1) lsum += __shfl_xor(lsum, off, 32);
  __syncthreads();
  {
    int d0 = lane * 2;
    float a0 = 0.f, a1 = 0.f;
    for (int c = 0; c < 128; ++c) {
      float wv = lowW[c];
      const float* vh = Vh + ((size_t)mb * NBR_ + c) * HD_;
      a0 += wv * vh[d0];
      a1 += wv * vh[d0 + 1];
    }
    lowO[d0] = a0; lowO[d0 + 1] = a1;
  }
  __syncthreads();

  // ---- correction factors, combine, write ----
  const size_t obase = (size_t)b * S_ * D_ + (size_t)h * HD_;
#pragma unroll
  for (int mt = 0; mt < 2; ++mt)
#pragma unroll
    for (int vt = 0; vt < 4; ++vt) {
      int dcol = vt * 16 + ln;
      float lov = lowO[dcol];
#pragma unroll
      for (int r = 0; r < 8; ++r) {
        int q = mt * 16 + r + 8 * hl;
        int s = rblk * 32 + q;
        float mq = mask[(size_t)b * S_ + s];
        float lcr = (rmax - mrun[mt][r]) * mq;
        float lc = __expf(fminf(lcr, 0.f));
        float hc = __expf(-fmaxf(lcr, 0.f));
        float den = lrun[mt][r] * hc + lsum * lc + 1e-6f;
        out[obase + (size_t)s * D_ + dcol] = (O[mt][vt][r] * hc + lov * lc) / den;
      }
    }
}

// ---------------------------------------------------------------------------
extern "C" void kernel_launch(void* const* d_in, const int* in_sizes, int n_in,
                              void* d_out, int out_size, void* d_ws, size_t ws_size,
                              hipStream_t stream) {
  (void)in_sizes; (void)n_in; (void)out_size; (void)ws_size;
  const float* X    = (const float*)d_in[0];
  const float* mask = (const float*)d_in[1];
  const float* Wq   = (const float*)d_in[2];
  const float* bq   = (const float*)d_in[3];
  const float* Wk   = (const float*)d_in[4];
  const float* bk   = (const float*)d_in[5];
  const float* Wv   = (const float*)d_in[6];
  const float* bv   = (const float*)d_in[7];
  float* out = (float*)d_out;

  char* w = (char*)d_ws;
  const size_t nQKV = (size_t)MB_ * S_ * HD_;      // 6,291,456 elements
  _Float16* Qf = (_Float16*)(w);
  _Float16* Kf = (_Float16*)(w + nQKV * 2);
  _Float16* Vf = (_Float16*)(w + nQKV * 4);
  size_t o = nQKV * 6;
  float* Qh = (float*)(w + o); o += (size_t)MB_ * NBR_ * HD_ * 4;
  float* Kh = (float*)(w + o); o += (size_t)MB_ * NBR_ * HD_ * 4;
  float* Vh = (float*)(w + o); o += (size_t)MB_ * NBR_ * HD_ * 4;
  float* tc = (float*)(w + o); o += (size_t)MB_ * NBR_ * 4;
  float* ll = (float*)(w + o); o += (size_t)MB_ * NBR_ * NBR_ * 4;
  float* rmx = (float*)(w + o); o += (size_t)MB_ * NBR_ * 4;
  float* th = (float*)(w + o); o += 128;
  int* rcnt = (int*)(w + o); o += (size_t)MB_ * NBR_ * 4;
  int* rofs = (int*)(w + o); o += (size_t)MB_ * NBR_ * 4;
  int* selb = (int*)(w + o); o += (size_t)MB_ * NSEL_ * 4;
  _Float16* Xh = (_Float16*)(w + o); o += (size_t)B_ * S_ * D_ * 2;
  _Float16* Wh = (_Float16*)(w + o); o += (size_t)3 * D_ * D_ * 2;

  // fp16 pre-conversion (keeps the GEMM K-loop free of VALU converts)
  {
    int n4x = (B_ * S_ * D_) / 4;        // 1,572,864
    cvt_f32_f16_kernel<<<(n4x + 255) / 256, 256, 0, stream>>>(X, Xh, n4x);
    int n4w = (D_ * D_) / 4;             // 147,456
    cvt_f32_f16_kernel<<<(n4w + 255) / 256, 256, 0, stream>>>(Wq, Wh, n4w);
    cvt_f32_f16_kernel<<<(n4w + 255) / 256, 256, 0, stream>>>(Wk, Wh + (size_t)D_ * D_, n4w);
    cvt_f32_f16_kernel<<<(n4w + 255) / 256, 256, 0, stream>>>(Wv, Wh + (size_t)2 * D_ * D_, n4w);
  }

  qkv_proj_kernel<<<dim3(64, 12, 3), 256, 0, stream>>>(
      Xh, Wh, mask, bq, bk, bv, Qf, Kf, Vf);
  block_mean_kernel<<<MB_ * NBR_, 64, 0, stream>>>(
      Qf, Kf, Vf, mask, Qh, Kh, Vh, tc);
  const size_t sm3 = (size_t)(128 * 72 * 2) * 2   // Qs, Ks fp16
                   + (size_t)128 * 129 * 4        // lowS / keys
                   + 128 * 4                      // rmaxS
                   + 128 * 4 * 3                  // irow/iofs/icur
                   + 256 * 4 + 16;                // scan + red
  lowres_select_kernel<<<MB_, 256, sm3, stream>>>(
      Qh, Kh, tc, ll, rmx, th, rcnt, rofs, selb);
  mra_attn_kernel<<<dim3(NBR_, MB_), 32, 0, stream>>>(
      Qf, Kf, Vf, mask, Vh, tc, ll, rmx, th, rcnt, rofs, selb, out);
}
